// FineTuneGNN_79834852098287
// MI455X (gfx1250) — compile-verified, matrix operand order checked by reading, not run on hardware
//
#include <hip/hip_runtime.h>

typedef __attribute__((ext_vector_type(16))) _Float16 v16h;
typedef __attribute__((ext_vector_type(8)))  float    v8f;

// ---------------- WMMA fragment helpers (wave32, ISA 7.12.2 layouts) ----------------

// A fragment: 16x32 f16 tile from row-major f32 matrix. base -> A[tile_row0][0], ld = row stride.
// lane&15 selects row; lanes 0-15 hold K = k0+0..7 and k0+16..23; lanes 16-31 hold K = k0+8..15 and k0+24..31.
__device__ __forceinline__ v16h frag_a_rowmajor(const float* __restrict__ base, int ld,
                                                int lane, int k0) {
    const float* p = base + (size_t)(lane & 15) * ld + k0 + ((lane >> 4) << 3);
    v16h f;
#pragma unroll
    for (int j = 0; j < 8; ++j) f[j] = (_Float16)p[j];
#pragma unroll
    for (int j = 0; j < 8; ++j) f[j + 8] = (_Float16)p[16 + j];
    return f;
}

// B fragment: 32x16 f16 tile from row-major f32 weight [K][ld]; column = col0 + (lane&15).
__device__ __forceinline__ v16h frag_b_rowmajor(const float* __restrict__ W, int ld,
                                                int lane, int k0, int col0) {
    const float* p = W + col0 + (lane & 15) + (size_t)(k0 + ((lane >> 4) << 3)) * ld;
    v16h f;
#pragma unroll
    for (int j = 0; j < 8; ++j) f[j] = (_Float16)p[(size_t)j * ld];
#pragma unroll
    for (int j = 0; j < 8; ++j) f[j + 8] = (_Float16)p[(size_t)(16 + j) * ld];
    return f;
}

// C/D store: lane&15 = column within tile; rows r + 8*(lane>=16).
__device__ __forceinline__ void store_c(float* __restrict__ O, int row0, int col, int ld,
                                        int lane, v8f c) {
    int rbase = row0 + ((lane >> 4) << 3);
#pragma unroll
    for (int r = 0; r < 8; ++r) O[(size_t)(rbase + r) * ld + col] = c[r];
}

// ---------------- kernels ----------------

__global__ void zero_kernel(float* p, long n) {
    long i = (long)blockIdx.x * blockDim.x + threadIdx.x;
    if (i < n) p[i] = 0.0f;
}

__global__ void deg_kernel(const int* __restrict__ dst, float* __restrict__ deg, int E) {
    int e = blockIdx.x * blockDim.x + threadIdx.x;
    if (e < E) atomicAdd(&deg[dst[e]], 1.0f);
}

// dinv[i] = rsqrt(deg_edges[i] + 1)   (+1 = self loop; deg >= 1 always)
__global__ void dinv_kernel(float* __restrict__ deg, int N) {
    int i = blockIdx.x * blockDim.x + threadIdx.x;
    if (i < N) deg[i] = rsqrtf(deg[i] + 1.0f);
}

// O[nrows x FOUT] = A[nrows x FIN] @ W[FIN x FOUT], f16 operands, f32 accumulate.
// One wave per 16-row tile; nrows must be a multiple of 16 (50000 = 3125*16).
template <int FIN, int FOUT>
__global__ void gemm_wmma(const float* __restrict__ A, const float* __restrict__ W,
                          float* __restrict__ O, int nrows) {
    int lane = threadIdx.x & 31;
    int wave = threadIdx.x >> 5;
    int tile = blockIdx.x * (blockDim.x >> 5) + wave;   // wave-uniform
    if (tile * 16 >= nrows) return;                      // whole-wave exit: EXEC stays all-ones
    int row0 = tile * 16;

    v8f acc[FOUT / 16] = {};
#pragma unroll
    for (int k0 = 0; k0 < FIN; k0 += 32) {
        v16h a = frag_a_rowmajor(A + (size_t)row0 * FIN, FIN, lane, k0);
#pragma unroll
        for (int t = 0; t < FOUT / 16; ++t) {
            v16h b = frag_b_rowmajor(W, FOUT, lane, k0, t * 16);
            acc[t] = __builtin_amdgcn_wmma_f32_16x16x32_f16(
                false, a, false, b, (short)0, acc[t], false, false);
        }
    }
#pragma unroll
    for (int t = 0; t < FOUT / 16; ++t)
        store_c(O, row0, t * 16 + (lane & 15), FOUT, lane, acc[t]);
}

// H[dst] += T[src] * dinv[src]*dinv[dst] over E real edges + N self loops.
// Thread layout: gid = edge*F + col -> lanes cover consecutive columns of one edge
// (coalesced gather of T, coalesced f32 atomics into H which lives in L2).
template <int F>
__global__ void scatter_edges(const float* __restrict__ T, const int* __restrict__ srcA,
                              const int* __restrict__ dstA, const float* __restrict__ dinv,
                              float* __restrict__ H, int E, int N) {
    long gid = (long)blockIdx.x * blockDim.x + threadIdx.x;
    long e = gid / F;
    int  c = (int)(gid & (F - 1));
    if (e >= (long)E + N) return;
    int s, d;
    if (e < E) { s = srcA[e]; d = dstA[e]; }
    else       { s = d = (int)(e - E); }
    float nrm = dinv[s] * dinv[d];
    atomicAdd(&H[(size_t)d * F + c], T[(size_t)s * F + c] * nrm);
}

template <int F, bool RELU>
__global__ void bias_act(float* __restrict__ H, const float* __restrict__ b, int total) {
    int i = blockIdx.x * blockDim.x + threadIdx.x;
    if (i < total) {
        float v = H[i] + b[i & (F - 1)];
        H[i] = RELU ? fmaxf(v, 0.0f) : v;
    }
}

// t3[i] = dot(h2[i,:32], W3[:,0])
__global__ void gemm3_kernel(const float* __restrict__ H2, const float* __restrict__ W3,
                             float* __restrict__ T3, int N) {
    int i = blockIdx.x * blockDim.x + threadIdx.x;
    if (i < N) {
        float s = 0.0f;
#pragma unroll
        for (int c = 0; c < 32; ++c) s += H2[(size_t)i * 32 + c] * W3[c];
        T3[i] = s;
    }
}

__global__ void scatter3_kernel(const int* __restrict__ srcA, const int* __restrict__ dstA,
                                const float* __restrict__ dinv, const float* __restrict__ T3,
                                float* __restrict__ H3, int E, int N) {
    int e = blockIdx.x * blockDim.x + threadIdx.x;
    if (e >= E + N) return;
    int s, d;
    if (e < E) { s = srcA[e]; d = dstA[e]; }
    else       { s = d = e - E; }
    atomicAdd(&H3[d], T3[s] * dinv[s] * dinv[d]);
}

// out[i][j] = (h3[i] + b3) * fcW[j] + fcb[j]
__global__ void final_kernel(const float* __restrict__ H3, const float* __restrict__ b3,
                             const float* __restrict__ fcW, const float* __restrict__ fcb,
                             float* __restrict__ out, int N) {
    int gid = blockIdx.x * blockDim.x + threadIdx.x;
    if (gid < N * 8) {
        int i = gid >> 3, j = gid & 7;
        out[gid] = (H3[i] + b3[0]) * fcW[j] + fcb[j];
    }
}

// ---------------- launcher ----------------

extern "C" void kernel_launch(void* const* d_in, const int* in_sizes, int n_in,
                              void* d_out, int out_size, void* d_ws, size_t ws_size,
                              hipStream_t stream) {
    const float* x   = (const float*)d_in[0];
    const int*   ei  = (const int*)d_in[1];   // [2, E]: row0 = src, row1 = dst
    const float* W1  = (const float*)d_in[2];
    const float* b1  = (const float*)d_in[3];
    const float* W2  = (const float*)d_in[4];
    const float* b2  = (const float*)d_in[5];
    const float* W3  = (const float*)d_in[6];
    const float* b3  = (const float*)d_in[7];
    const float* fcW = (const float*)d_in[8];
    const float* fcb = (const float*)d_in[9];
    float* out = (float*)d_out;

    const int N = in_sizes[0] / 128;
    const int E = in_sizes[1] / 2;
    const int* srcA = ei;
    const int* dstA = ei + E;

    // workspace layout (floats): [dinv N][h1 N*64][h2 N*32][h3 N][t1 N*64][t2 N*32][t3 N]
    float* ws   = (float*)d_ws;
    float* dinv = ws;
    float* h1   = dinv + N;
    float* h2   = h1 + (size_t)N * 64;
    float* h3   = h2 + (size_t)N * 32;
    float* t1   = h3 + N;
    float* t2   = t1 + (size_t)N * 64;
    float* t3   = t2 + (size_t)N * 32;

    const int B = 256;

    // zero atomic targets: dinv + h1 + h2 + h3 = N*(1+64+32+1) contiguous floats
    long nz = (long)N * 98;
    zero_kernel<<<(unsigned)((nz + B - 1) / B), B, 0, stream>>>(ws, nz);

    deg_kernel<<<(E + B - 1) / B, B, 0, stream>>>(dstA, dinv, E);
    dinv_kernel<<<(N + B - 1) / B, B, 0, stream>>>(dinv, N);

    int tiles  = N / 16;               // 50000 -> 3125 full tiles
    int gblk   = (tiles + 7) / 8;      // 8 waves / block

    // layer 1: 128 -> 64
    gemm_wmma<128, 64><<<gblk, B, 0, stream>>>(x, W1, t1, N);
    long th1 = ((long)E + N) * 64;
    scatter_edges<64><<<(unsigned)((th1 + B - 1) / B), B, 0, stream>>>(t1, srcA, dstA, dinv, h1, E, N);
    bias_act<64, true><<<(N * 64 + B - 1) / B, B, 0, stream>>>(h1, b1, N * 64);

    // layer 2: 64 -> 32
    gemm_wmma<64, 32><<<gblk, B, 0, stream>>>(h1, W2, t2, N);
    long th2 = ((long)E + N) * 32;
    scatter_edges<32><<<(unsigned)((th2 + B - 1) / B), B, 0, stream>>>(t2, srcA, dstA, dinv, h2, E, N);
    bias_act<32, true><<<(N * 32 + B - 1) / B, B, 0, stream>>>(h2, b2, N * 32);

    // layer 3: 32 -> 1 + aggregation, then 1 -> 8 fc (b3 folded in)
    gemm3_kernel<<<(N + B - 1) / B, B, 0, stream>>>(h2, W3, t3, N);
    scatter3_kernel<<<(E + N + B - 1) / B, B, 0, stream>>>(srcA, dstA, dinv, t3, h3, E, N);
    final_kernel<<<(N * 8 + B - 1) / B, B, 0, stream>>>(h3, b3, fcW, fcb, out, N);
}
